// NF4_34935263985944
// MI455X (gfx1250) — matrix operand order repaired and checked by reading, not compiled, not run on hardware
//
#include <hip/hip_runtime.h>

// ---------------------------------------------------------------------------
// Fused NF4-dequant double GEMM for MI455X (gfx1250, wave32, WMMA bf16)
//   out[32768,512] = (x[32768,512] @ w_lora^T) @ W^T
// Weights pre-packed into WMMA B-fragment order (bf16) in d_ws.
// MTILE=64: each wave holds 4 M-subtiles x 4 N-subtiles of accumulators, so
// every B fragment load feeds 4 WMMAs (4x less L2 weight traffic than naive).
// y is held in registers between the GEMMs and written back into the same
// 64 KB LDS tile (bf16) after a barrier.
// ---------------------------------------------------------------------------

#define D       512
#define MTILE   64
#define KSTEPS  16           // 512 / 32 (bf16 WMMA K = 32)
#define PACK_DWORDS 131072   // 512*512/2 dwords per packed weight matrix

typedef __attribute__((ext_vector_type(16))) __bf16 v16bf;
typedef __attribute__((ext_vector_type(8)))  float  v8f;

struct U128 { unsigned x, y, z, w; };
struct U256 { U128 lo, hi; };
union  FragU { U256 u; v16bf v; };
struct U64x { unsigned lo, hi; };
struct I64x { int a, b; };
struct F4x  { float a, b, c, d; };

__constant__ float c_nf4[16] = {
    -1.0f, -0.6961928009986877f, -0.5250730514526367f, -0.39491748809814453f,
    -0.28444138169288635f, -0.18477343022823334f, -0.09105003625154495f, 0.0f,
    0.07958029955625534f, 0.16093020141124725f, 0.24611230194568634f,
    0.33791524171829224f, 0.44070982933044434f, 0.5626170039176941f,
    0.7229568362236023f, 1.0f };

__device__ __forceinline__ unsigned short f2bf(float f) {
    unsigned u = __float_as_uint(f);
    return (unsigned short)((u + 0x7FFFu + ((u >> 16) & 1u)) >> 16);
}

// ---------------------------------------------------------------------------
// Pack kernel 1: NF4 dequant -> bf16, swizzled into WMMA B-fragment layout.
// B-tile (kt,nt) covers K=[kt*32,kt*32+32) x N=[nt*16,nt*16+16).
// dword index = (kt*32+nt)*256 + lane*8 + j
//   lane = (N%16) + 16*Kgroup , dword j holds K = Kgroup*16 + 2j, 2j+1
// B[k][n] = dequant(qidx[n][k])  (because y = x @ w_lora^T)
// ---------------------------------------------------------------------------
__global__ void nf4_pack_kernel(const int* __restrict__ qidx,
                                const float* __restrict__ scales,
                                unsigned* __restrict__ bpack) {
    int t    = blockIdx.x * blockDim.x + threadIdx.x;   // 0 .. 131071
    int j    = t & 7;
    int l    = (t >> 3) & 31;
    int tile = t >> 8;
    int nt   = tile & 31;
    int kt   = tile >> 5;
    int n    = nt * 16 + (l & 15);
    int k    = kt * 32 + (l >> 4) * 16 + 2 * j;

    I64x q  = *(const I64x*)(qidx + (size_t)n * D + k);
    float s = scales[((size_t)n * D + k) >> 6];         // same 64-block for k,k+1
    unsigned short lo = f2bf(c_nf4[q.a & 15] * s);
    unsigned short hi = f2bf(c_nf4[q.b & 15] * s);
    bpack[t] = (unsigned)lo | ((unsigned)hi << 16);
}

// Pack kernel 2: dense W fp32 -> bf16, same fragment swizzle. B[k][n] = W[n][k].
__global__ void w_pack_kernel(const float* __restrict__ W,
                              unsigned* __restrict__ bpack) {
    int t    = blockIdx.x * blockDim.x + threadIdx.x;
    int j    = t & 7;
    int l    = (t >> 3) & 31;
    int tile = t >> 8;
    int nt   = tile & 31;
    int kt   = tile >> 5;
    int n    = nt * 16 + (l & 15);
    int k    = kt * 32 + (l >> 4) * 16 + 2 * j;

    float a = W[(size_t)n * D + k];
    float b = W[(size_t)n * D + k + 1];
    bpack[t] = (unsigned)f2bf(a) | ((unsigned)f2bf(b) << 16);
}

// ---------------------------------------------------------------------------
// Fused kernel: per WG, 64 rows of x in one 64 KB bf16 LDS tile.
// 8 waves; wave w owns all 4 M-subtiles and N-subtiles [w*4, w*4+4).
// ---------------------------------------------------------------------------
__global__ void __launch_bounds__(256)
nf4_fused_gemm_kernel(const float* __restrict__ x,
                      const unsigned* __restrict__ bp1,
                      const unsigned* __restrict__ bp2,
                      float* __restrict__ out) {
    __shared__ __align__(16) unsigned char lds[65536];   // 64x512 bf16 tile
    unsigned short* xs = (unsigned short*)lds;

    const int tid  = threadIdx.x;
    const int wg   = blockIdx.x;
    const int lane = tid & 31;
    const int wave = tid >> 5;

    // ---- stage 0: load x tile, convert fp32 -> bf16 into LDS --------------
    const float* xg = x + (size_t)wg * (MTILE * D);
    #pragma unroll 4
    for (int it = 0; it < 32; ++it) {
        int v = it * 256 + tid;                  // float4 index, coalesced
        F4x f = *(const F4x*)(xg + (size_t)v * 4);
        U64x p;
        p.lo = (unsigned)f2bf(f.a) | ((unsigned)f2bf(f.b) << 16);
        p.hi = (unsigned)f2bf(f.c) | ((unsigned)f2bf(f.d) << 16);
        *(U64x*)((char*)xs + (size_t)v * 8) = p;
    }
    __syncthreads();

    const int ntbase = wave * 4;          // 4 N-subtiles per wave
    const int row    = lane & 15;
    const int g      = lane >> 4;
    const char* xsb  = (const char*)xs;

    // per-lane A fragment base offsets for the 4 M-subtiles
    // (contiguous A layout matching B pack: lanes 0-15 K 0..15 ; lanes 16-31 K 16..31)
    const size_t arow = (size_t)row * (D * 2) + (size_t)g * 32;

    // ---- GEMM1: y = x_tile @ w_lora^T -------------------------------------
    v8f acc[4][4] = {};
    for (int kt = 0; kt < KSTEPS; ++kt) {
        FragU A[4];
        #pragma unroll
        for (int mi = 0; mi < 4; ++mi) {
            size_t aoff = arow + (size_t)mi * (16 * D * 2) + (size_t)kt * 64;
            A[mi].u.lo = *(const U128*)(xsb + aoff);
            A[mi].u.hi = *(const U128*)(xsb + aoff + 16);
        }
        #pragma unroll
        for (int ni = 0; ni < 4; ++ni) {
            FragU Bf;
            const char* bb = (const char*)bp1 +
                             ((size_t)(kt * 32 + ntbase + ni) * 1024) + (size_t)lane * 32;
            Bf.u.lo = *(const U128*)(bb);
            Bf.u.hi = *(const U128*)(bb + 16);
            #pragma unroll
            for (int mi = 0; mi < 4; ++mi)
                acc[mi][ni] = __builtin_amdgcn_wmma_f32_16x16x32_bf16(
                    false, A[mi].v, false, Bf.v, (short)0, acc[mi][ni], false, false);
        }
    }

    // all waves done reading x tile; overwrite it in place with bf16 y
    __syncthreads();
    #pragma unroll
    for (int mi = 0; mi < 4; ++mi) {
        #pragma unroll
        for (int ni = 0; ni < 4; ++ni) {
            int col = (ntbase + ni) * 16 + row;
            #pragma unroll
            for (int r = 0; r < 8; ++r) {
                int rr = mi * 16 + g * 8 + r;          // C layout: M = g*8 + r
                xs[rr * D + col] = f2bf(acc[mi][ni][r]);
            }
        }
    }
    __syncthreads();

    // ---- GEMM2: out = y_tile @ W^T ----------------------------------------
    v8f acc2[4][4] = {};
    for (int kt = 0; kt < KSTEPS; ++kt) {
        FragU A[4];
        #pragma unroll
        for (int mi = 0; mi < 4; ++mi) {
            size_t aoff = arow + (size_t)mi * (16 * D * 2) + (size_t)kt * 64;
            A[mi].u.lo = *(const U128*)(xsb + aoff);
            A[mi].u.hi = *(const U128*)(xsb + aoff + 16);
        }
        #pragma unroll
        for (int ni = 0; ni < 4; ++ni) {
            FragU Bf;
            const char* bb = (const char*)bp2 +
                             ((size_t)(kt * 32 + ntbase + ni) * 1024) + (size_t)lane * 32;
            Bf.u.lo = *(const U128*)(bb);
            Bf.u.hi = *(const U128*)(bb + 16);
            #pragma unroll
            for (int mi = 0; mi < 4; ++mi)
                acc2[mi][ni] = __builtin_amdgcn_wmma_f32_16x16x32_bf16(
                    false, A[mi].v, false, Bf.v, (short)0, acc2[mi][ni], false, false);
        }
    }

    float* og = out + (size_t)wg * (MTILE * D);
    #pragma unroll
    for (int mi = 0; mi < 4; ++mi) {
        #pragma unroll
        for (int ni = 0; ni < 4; ++ni) {
            int col = (ntbase + ni) * 16 + row;
            #pragma unroll
            for (int r = 0; r < 8; ++r) {
                int rr = mi * 16 + g * 8 + r;
                og[rr * D + col] = acc2[mi][ni][r];
            }
        }
    }
}

// ---------------------------------------------------------------------------
extern "C" void kernel_launch(void* const* d_in, const int* in_sizes, int n_in,
                              void* d_out, int out_size, void* d_ws, size_t ws_size,
                              hipStream_t stream) {
    const float* x      = (const float*)d_in[0];   // [4,8192,512] fp32
    const int*   qidx   = (const int*)  d_in[1];   // [512,512] int32
    const float* scales = (const float*)d_in[2];   // [4096] fp32
    const float* W      = (const float*)d_in[3];   // [512,512] fp32

    unsigned* bp1 = (unsigned*)d_ws;               // 512 KB packed w_lora (bf16)
    unsigned* bp2 = bp1 + PACK_DWORDS;             // 512 KB packed W      (bf16)

    nf4_pack_kernel<<<PACK_DWORDS / 256, 256, 0, stream>>>(qidx, scales, bp1);
    w_pack_kernel  <<<PACK_DWORDS / 256, 256, 0, stream>>>(W, bp2);

    const int M = 4 * 8192;                        // 32768 rows
    nf4_fused_gemm_kernel<<<M / MTILE, 256, 0, stream>>>(x, bp1, bp2, (float*)d_out);
}
